// ContinuousConvolutionModel_24429773980232
// MI455X (gfx1250) — matrix operand order repaired and checked by reading, not compiled, not run on hardware
//
#include <hip/hip_runtime.h>
#include <hip/hip_bf16.h>

// ---------------------------------------------------------------------------
// ContinuousConv model for MI455X (gfx1250, wave32, WMMA).
// Per-layer fused: geometry -> LDS f32 scatter -> one-shot f32->f16 convert
// pass -> all-8-wave software-pipelined WMMA f16 GEMM (K-split) vs
// L2-resident f16 weights.
// ---------------------------------------------------------------------------

typedef _Float16 v16h __attribute__((ext_vector_type(16)));
typedef _Float16 v8h  __attribute__((ext_vector_type(8)));
typedef float    v8f  __attribute__((ext_vector_type(8)));

#define KNB    32
#define KS     6
#define MTOT   216            // 6*6*6
#define TP     16             // points per tile (WMMA M)
#define PAIRS  (TP * KNB)     // 512 (point, neighbor) pairs per tile

// LDS layout (geometry region, before the A-chunks):
//   s_cell : u16 [PAIRS*8]   @ 0      (8192 B)
//   s_w    : f32 [PAIRS*8]   @ 8192   (16384 B)
//   s_nb   : i32 [PAIRS]     @ 24576  (2048 B)
//   s_cnt  : f32 [TP]        @ 26624  (64 B)
//   s_Af32 : f32 [TP*MC*CIN] @ 26688           (also reused for k-split reduce)
//   s_Ah   : f16 [TP*MC*CIN] @ 26688 + 4*TP*MC*CIN
#define SMEM_GEO 26688

// ------------------------- geometry helpers --------------------------------

__device__ __forceinline__ void ball_to_cube(float x, float y, float z,
                                             float& xq, float& yq, float& zq) {
    const float EPSF = 1e-12f;
    const float FOPI = 1.27323954473516f;  // 4/pi
    float sq   = x * x + y * y + z * z;
    float nrm  = sqrtf(fmaxf(sq, EPSF));
    float rho2 = x * x + y * y;
    bool  in_cone = 1.25f * z * z > rho2;
    float s1 = sqrtf(3.0f * nrm / (nrm + fabsf(z)));
    float s2 = nrm * rsqrtf(fmaxf(rho2, EPSF));
    float s  = in_cone ? s1 : s2;
    float xc = x * s;
    float yc = y * s;
    float zc = in_cone ? copysignf(nrm, z) : 1.5f * z;
    if (sq < EPSF) { xc = 0.f; yc = 0.f; zc = 0.f; }
    // cylinder -> cube (inverse concentric map on the xy disc)
    float sq_xy = xc * xc + yc * yc;
    float nxy   = sqrtf(fmaxf(sq_xy, EPSF));
    bool  x_major = fabsf(yc) <= fabsf(xc);
    float xd = (fabsf(xc) < EPSF) ? 1.0f : xc;
    float yd = (fabsf(yc) < EPSF) ? 1.0f : yc;
    float tx = (xc == 0.f) ? 0.f : copysignf(nxy, xc);
    float ty = (yc == 0.f) ? 0.f : copysignf(nxy, yc);
    float xo = x_major ? tx : ty * FOPI * atanf(xc / yd);
    float yo = x_major ? tx * FOPI * atanf(yc / xd) : ty;
    if (sq_xy < EPSF) { xo = 0.f; yo = 0.f; }
    xq = xo; yq = yo; zq = zc;
}

__device__ __forceinline__ void axis_setup(float coord, int& a0, int& a1, float& f) {
    float i0f = floorf(coord);
    f = coord - i0f;
    int i0 = (int)i0f;
    i0 = min(max(i0, 0), KS - 1);
    a0 = i0;
    a1 = min(i0 + 1, KS - 1);
}

// ------------------------- weight transpose/convert ------------------------

// W [MF, cout] f32 -> Wt [cout, MF] f16   (MF = 216*Cin, contraction-contig)
__global__ void wtrans_kernel(const float* __restrict__ W,
                              _Float16* __restrict__ Wt, int MF, int cout) {
    int i = blockIdx.x * blockDim.x + threadIdx.x;
    int total = MF * cout;
    if (i < total) {
        int co = i / MF;
        int mf = i - co * MF;
        Wt[i] = (_Float16)W[mf * cout + co];
    }
}

// W [cin, cout_real] f32 -> Wt [cout_pad, cin] f16, zero-padded columns
__global__ void fctrans_kernel(const float* __restrict__ W,
                               _Float16* __restrict__ Wt,
                               int cin, int cout_real, int cout_pad) {
    int i = blockIdx.x * blockDim.x + threadIdx.x;
    int total = cout_pad * cin;
    if (i < total) {
        int co = i / cin;
        int ci = i - co * cin;
        Wt[i] = (co < cout_real) ? (_Float16)W[ci * cout_real + co] : (_Float16)0.0f;
    }
}

// ------------------------- fused continuous-conv layer ---------------------

template <int CIN, int COUT, int MC>
__global__ __launch_bounds__(256) void conv_kernel(
    const float* __restrict__ xin,          // [n, CIN]
    const float* __restrict__ pos,          // [n, 3]
    const int* __restrict__ nidx,           // [n, KNB]
    const unsigned char* __restrict__ nmask,// [n, KNB]
    const _Float16* __restrict__ Wt,        // [COUT][MTOT*CIN] f16
    const float* __restrict__ bias,         // [COUT]
    float* __restrict__ xout,               // [n, COUT]
    int n) {
    constexpr int MFTOT  = MTOT * CIN;
    constexpr int ACHUNK = TP * MC * CIN;       // floats in one A chunk
    constexpr int NCOT   = COUT / 16;           // co-tiles (waves per k-group)
    constexpr int KSPLIT = 8 / NCOT;            // k-split factor (all 8 waves busy)
    constexpr int KSTEPS = (MC * CIN) / 32;     // WMMA k-steps per chunk

    extern __shared__ char smem[];
    unsigned short* s_cell = (unsigned short*)(smem);
    float*          s_w    = (float*)(smem + 8192);
    int*            s_nb   = (int*)(smem + 24576);
    float*          s_cnt  = (float*)(smem + 26624);
    float*          s_Af32 = (float*)(smem + SMEM_GEO);
    _Float16*       s_Ah   = (_Float16*)(smem + SMEM_GEO + ACHUNK * 4);
    float*          s_red  = (float*)(smem + SMEM_GEO);   // reuse after chunks

    const int tid = threadIdx.x;
    if (tid < TP) s_cnt[tid] = 0.0f;
    __syncthreads();

    // ---- Stage 1: geometry (trilinear corners + weights) into LDS ----
    for (int p = tid; p < PAIRS; p += 256) {
        int j  = p >> 5;            // point within tile
        int k  = p & (KNB - 1);     // neighbor
        int gj = blockIdx.x * TP + j;
        if (gj < n) {
            int   ni = nidx[gj * KNB + k];
            float mf = nmask[gj * KNB + k] ? 1.0f : 0.0f;
            s_nb[p] = ni;
            float ux = (pos[ni * 3 + 0] - pos[gj * 3 + 0]) * (2.0f / 3.0f);
            float uy = (pos[ni * 3 + 1] - pos[gj * 3 + 1]) * (2.0f / 3.0f);
            float uz = (pos[ni * 3 + 2] - pos[gj * 3 + 2]) * (2.0f / 3.0f);
            float cx, cy, cz;
            ball_to_cube(ux, uy, uz, cx, cy, cz);
            cx = (cx + 1.0f) * (0.5f * (KS - 1));   // align_corners
            cy = (cy + 1.0f) * (0.5f * (KS - 1));
            cz = (cz + 1.0f) * (0.5f * (KS - 1));
            int ix0, ix1, iy0, iy1, iz0, iz1;
            float fx, fy, fz;
            axis_setup(cx, ix0, ix1, fx);
            axis_setup(cy, iy0, iy1, fy);
            axis_setup(cz, iz0, iz1, fz);
            float wx[2] = {1.0f - fx, fx};
            float wy[2] = {1.0f - fy, fy};
            float wz[2] = {1.0f - fz, fz};
            int   axx[2] = {ix0, ix1}, ayy[2] = {iy0, iy1}, azz[2] = {iz0, iz1};
#pragma unroll
            for (int a = 0; a < 2; ++a)
#pragma unroll
                for (int b = 0; b < 2; ++b)
#pragma unroll
                    for (int c = 0; c < 2; ++c) {
                        int idx = ((a << 2) | (b << 1) | c);
                        s_w[p * 8 + idx] = wx[a] * wy[b] * wz[c] * mf;
                        s_cell[p * 8 + idx] =
                            (unsigned short)((axx[a] * KS + ayy[b]) * KS + azz[c]);
                    }
            if (mf > 0.0f) atomicAdd(&s_cnt[j], 1.0f);
        } else {
            s_nb[p] = 0;
#pragma unroll
            for (int c = 0; c < 8; ++c) { s_w[p * 8 + c] = 0.0f; s_cell[p * 8 + c] = 0; }
        }
    }
    __syncthreads();

    // ---- Stage 2: chunked scatter -> f16 convert -> all-wave WMMA ----
    const int wv   = tid >> 5;        // wave id (0..7)
    const int lane = tid & 31;
    const int m    = lane & 15;       // A row / B col within 16x16 tile
    const int half = lane >> 4;       // WMMA K-half selector
    const int cot  = wv % NCOT;       // co-tile owned by this wave
    const int ksl  = wv / NCOT;       // k-split group
    v8f acc;
#pragma unroll
    for (int r = 0; r < 8; ++r) acc[r] = 0.0f;

    const _Float16* wbase0 =
        Wt + (size_t)(cot * 16 + m) * MFTOT + half * 16;
    const _Float16* abase = s_Ah + m * (MC * CIN);

    for (int c0 = 0; c0 < MTOT; c0 += MC) {
        // zero the f32 A-chunk
        for (int i = tid; i < ACHUNK; i += 256) s_Af32[i] = 0.0f;
        __syncthreads();
        // scatter-add trilinear-weighted neighbor features (ds_add_f32)
        for (int tt = tid; tt < PAIRS * 8; tt += 256) {
            float w = s_w[tt];
            if (w != 0.0f) {
                int p = tt >> 3;
                unsigned rel = (unsigned)((int)s_cell[tt] - c0);
                if (rel < (unsigned)MC) {
                    const float4* frow =
                        (const float4*)(xin + (size_t)s_nb[p] * CIN);
                    float* arow = s_Af32 + (p >> 5) * (MC * CIN) + (int)rel * CIN;
#pragma unroll
                    for (int q = 0; q < CIN / 4; ++q) {
                        float4 v = frow[q];
                        atomicAdd(arow + 4 * q + 0, w * v.x);
                        atomicAdd(arow + 4 * q + 1, w * v.y);
                        atomicAdd(arow + 4 * q + 2, w * v.z);
                        atomicAdd(arow + 4 * q + 3, w * v.w);
                    }
                }
            }
        }
        __syncthreads();
        // one-shot f32 -> f16 conversion of the whole chunk (block-wide)
        for (int i = tid * 8; i < ACHUNK; i += 256 * 8) {
            const float4* src = (const float4*)(s_Af32 + i);
            float4 a4 = src[0], b4 = src[1];
            _Float16* dst = s_Ah + i;
            dst[0] = (_Float16)a4.x; dst[1] = (_Float16)a4.y;
            dst[2] = (_Float16)a4.z; dst[3] = (_Float16)a4.w;
            dst[4] = (_Float16)b4.x; dst[5] = (_Float16)b4.y;
            dst[6] = (_Float16)b4.z; dst[7] = (_Float16)b4.w;
        }
        __syncthreads();
        // Software-pipelined WMMA chain: every wave owns (co-tile, k-slice).
        // Fragment loads for step s+1 are issued BEFORE the WMMA of step s so
        // VMEM/LDS latency overlaps the matrix pipe (partial s_wait counts).
        {
            const _Float16* wbase = wbase0 + (size_t)c0 * CIN;
            const int s0 = (ksl * KSTEPS) / KSPLIT;
            const int s1 = ((ksl + 1) * KSTEPS) / KSPLIT;
            // A frag: lanes 0-15 hold K {0..15}; lanes 16-31 {16..31}
            v8h  lo0 = *(const v8h*)(abase + s0 * 32 + 16 * half);
            v8h  hi0 = *(const v8h*)(abase + s0 * 32 + 8 + 16 * half);
            v16h b0  = *(const v16h*)(wbase + s0 * 32);
            for (int s = s0; s + 1 < s1; ++s) {
                v8h  lo1 = *(const v8h*)(abase + (s + 1) * 32 + 16 * half);
                v8h  hi1 = *(const v8h*)(abase + (s + 1) * 32 + 8 + 16 * half);
                v16h b1  = *(const v16h*)(wbase + (s + 1) * 32);
                __builtin_prefetch(wbase + (s + 1) * 32 + 512, 0, 3);
                v16h a;
#pragma unroll
                for (int i = 0; i < 8; ++i) { a[i] = lo0[i]; a[i + 8] = hi0[i]; }
                acc = __builtin_amdgcn_wmma_f32_16x16x32_f16(
                    false, a, false, b0, (short)0, acc, false, false);
                lo0 = lo1; hi0 = hi1; b0 = b1;
            }
            v16h a;
#pragma unroll
            for (int i = 0; i < 8; ++i) { a[i] = lo0[i]; a[i + 8] = hi0[i]; }
            acc = __builtin_amdgcn_wmma_f32_16x16x32_f16(
                false, a, false, b0, (short)0, acc, false, false);
        }
        __syncthreads();
    }

    // ---- k-split reduction through LDS (A-chunk region is free now) ----
    if (ksl > 0) {
        float* dst = s_red + (((ksl - 1) * NCOT + cot) * 32 + lane) * 8;
#pragma unroll
        for (int r = 0; r < 8; ++r) dst[r] = acc[r];
    }
    __syncthreads();
    if (ksl == 0) {
#pragma unroll
        for (int g = 1; g < KSPLIT; ++g) {
            const float* src = s_red + (((g - 1) * NCOT + cot) * 32 + lane) * 8;
#pragma unroll
            for (int r = 0; r < 8; ++r) acc[r] += src[r];
        }
        // ---- Epilogue: normalize by neighbor count, bias, ReLU ----
        int col = cot * 16 + m;
        float bv = bias[col];
#pragma unroll
        for (int r = 0; r < 8; ++r) {
            int row = r + 8 * half;   // C/D layout: VGPR r -> M=r / r+8
            int gj = blockIdx.x * TP + row;
            if (gj < n) {
                float cnt = s_cnt[row];
                float v = acc[r];
                if (cnt > 0.0f) v /= cnt;
                v = fmaxf(v + bv, 0.0f);
                xout[(size_t)gj * COUT + col] = v;
            }
        }
    }
}

// ------------------------- dense FC layer (WMMA) ---------------------------

template <int CIN, int COUT_PAD, bool RELU>
__global__ __launch_bounds__(32) void fc_kernel(
    const float* __restrict__ xin,    // [n, CIN]
    const _Float16* __restrict__ Wt,  // [COUT_PAD][CIN] f16
    const float* __restrict__ bias,   // [cout_store]
    float* __restrict__ xout,         // [n, cout_store]
    int n, int cout_store) {
    const int lane = threadIdx.x;
    const int m    = lane & 15;
    const int half = lane >> 4;
    const int row0 = blockIdx.x * TP;
    const int cot  = blockIdx.y;
    v8f acc;
#pragma unroll
    for (int r = 0; r < 8; ++r) acc[r] = 0.0f;

    int arow = min(row0 + m, n - 1);   // clamp (garbage rows never stored)
    const float* xr = xin + (size_t)arow * CIN;
    const _Float16* wr = Wt + (size_t)(cot * 16 + m) * CIN + half * 16;
#pragma unroll
    for (int s = 0; s < CIN / 32; ++s) {
        v16h a;
#pragma unroll
        for (int i = 0; i < 8; ++i) {
            a[i]     = (_Float16)xr[s * 32 + 16 * half + i];
            a[i + 8] = (_Float16)xr[s * 32 + 8 + 16 * half + i];
        }
        v16h b = *(const v16h*)(wr + s * 32);
        acc = __builtin_amdgcn_wmma_f32_16x16x32_f16(
            false, a, false, b, (short)0, acc, false, false);
    }
    int col = cot * 16 + m;
    if (col < cout_store) {
        float bv = bias[col];
#pragma unroll
        for (int r = 0; r < 8; ++r) {
            int row = row0 + r + 8 * half;
            if (row < n) {
                float v = acc[r] + bv;
                if (RELU) v = fmaxf(v, 0.0f);
                xout[(size_t)row * cout_store + col] = v;
            }
        }
    }
}

// ------------------------- host-side launch --------------------------------

extern "C" void kernel_launch(void* const* d_in, const int* in_sizes, int n_in,
                              void* d_out, int out_size, void* d_ws, size_t ws_size,
                              hipStream_t stream) {
    const float* feats = (const float*)d_in[0];
    const float* pos   = (const float*)d_in[1];
    const int*   nidx  = (const int*)d_in[2];
    const unsigned char* nmask = (const unsigned char*)d_in[3];
    const float* W1 = (const float*)d_in[4];  const float* b1 = (const float*)d_in[5];
    const float* W2 = (const float*)d_in[6];  const float* b2 = (const float*)d_in[7];
    const float* W3 = (const float*)d_in[8];  const float* b3 = (const float*)d_in[9];
    const float* Wfc1 = (const float*)d_in[10]; const float* bfc1 = (const float*)d_in[11];
    const float* Wfc2 = (const float*)d_in[12]; const float* bfc2 = (const float*)d_in[13];
    const float* Wfc3 = (const float*)d_in[14]; const float* bfc3 = (const float*)d_in[15];
    const float* Wout = (const float*)d_in[16]; const float* bout = (const float*)d_in[17];

    const int n = in_sizes[0] / 4;  // feats is [n, 4]

    // workspace carve-out (256B aligned)
    char*  ws = (char*)d_ws;
    size_t off = 0;
    auto carve = [&](size_t bytes) -> char* {
        char* p = ws + off;
        off = (off + bytes + 255) & ~(size_t)255;
        return p;
    };
    _Float16* Wt1 = (_Float16*)carve((size_t)MTOT * 4 * 64 * 2);
    _Float16* Wt2 = (_Float16*)carve((size_t)MTOT * 64 * 64 * 2);
    _Float16* Wt3 = (_Float16*)carve((size_t)MTOT * 64 * 32 * 2);
    _Float16* Ft1 = (_Float16*)carve(64 * 32 * 2);
    _Float16* Ft2 = (_Float16*)carve(64 * 64 * 2);
    _Float16* Ft3 = (_Float16*)carve(32 * 64 * 2);
    _Float16* Fto = (_Float16*)carve(16 * 32 * 2);
    float* xA = (float*)carve((size_t)n * 64 * 4);
    float* xB = (float*)carve((size_t)n * 64 * 4);

    // ---- weight transpose/convert to f16 ----
    {
        int t1 = MTOT * 4 * 64, t2 = MTOT * 64 * 64, t3 = MTOT * 64 * 32;
        wtrans_kernel<<<(t1 + 255) / 256, 256, 0, stream>>>(W1, Wt1, MTOT * 4, 64);
        wtrans_kernel<<<(t2 + 255) / 256, 256, 0, stream>>>(W2, Wt2, MTOT * 64, 64);
        wtrans_kernel<<<(t3 + 255) / 256, 256, 0, stream>>>(W3, Wt3, MTOT * 64, 32);
        fctrans_kernel<<<(64 * 32 + 255) / 256, 256, 0, stream>>>(Wfc1, Ft1, 32, 64, 64);
        fctrans_kernel<<<(64 * 64 + 255) / 256, 256, 0, stream>>>(Wfc2, Ft2, 64, 64, 64);
        fctrans_kernel<<<(32 * 64 + 255) / 256, 256, 0, stream>>>(Wfc3, Ft3, 64, 32, 32);
        fctrans_kernel<<<(16 * 32 + 255) / 256, 256, 0, stream>>>(Wout, Fto, 32, 3, 16);
    }

    const int nblk = (n + TP - 1) / TP;

    // conv layers: dynamic LDS = geo + f32 chunk + f16 chunk
    size_t smem1 = SMEM_GEO + (size_t)TP * 216 * 4 * 6;   // 4B + 2B per element
    size_t smem2 = SMEM_GEO + (size_t)TP * 18 * 64 * 6;
    (void)hipFuncSetAttribute((const void*)conv_kernel<4, 64, 216>,
                              hipFuncAttributeMaxDynamicSharedMemorySize, (int)smem1);
    (void)hipFuncSetAttribute((const void*)conv_kernel<64, 64, 18>,
                              hipFuncAttributeMaxDynamicSharedMemorySize, (int)smem2);
    (void)hipFuncSetAttribute((const void*)conv_kernel<64, 32, 18>,
                              hipFuncAttributeMaxDynamicSharedMemorySize, (int)smem2);

    conv_kernel<4, 64, 216><<<nblk, 256, smem1, stream>>>(
        feats, pos, nidx, nmask, Wt1, b1, xA, n);
    conv_kernel<64, 64, 18><<<nblk, 256, smem2, stream>>>(
        xA, pos, nidx, nmask, Wt2, b2, xB, n);
    conv_kernel<64, 32, 18><<<nblk, 256, smem2, stream>>>(
        xB, pos, nidx, nmask, Wt3, b3, xA, n);

    // FC stack
    fc_kernel<32, 64, true><<<dim3(nblk, 4), 32, 0, stream>>>(xA, Ft1, bfc1, xB, n, 64);
    fc_kernel<64, 64, true><<<dim3(nblk, 4), 32, 0, stream>>>(xB, Ft2, bfc2, xA, n, 64);
    fc_kernel<64, 32, true><<<dim3(nblk, 2), 32, 0, stream>>>(xA, Ft3, bfc3, xB, n, 32);
    fc_kernel<32, 16, false><<<dim3(nblk, 1), 32, 0, stream>>>(xB, Fto, bout,
                                                               (float*)d_out, n, 3);
}
